// QRNN_45887430590621
// MI455X (gfx1250) — matrix operand order repaired and checked by reading, not compiled, not run on hardware
//
#include <hip/hip_runtime.h>

// ---------------------------------------------------------------------------
// QRNN for MI455X (gfx1250): bf16 WMMA GEMM for the 3 causal convs with
// async-to-LDS weight staging, fused activations, then a bandwidth-bound
// sequential fo-pool scan.
// ---------------------------------------------------------------------------

typedef __bf16 bf16;
typedef bf16 v16bf __attribute__((ext_vector_type(16)));
typedef bf16 v8bf  __attribute__((ext_vector_type(8)));
typedef float v8f  __attribute__((ext_vector_type(8)));
typedef int   v4i  __attribute__((ext_vector_type(4)));

#define Bx 8
#define Tx 4096
#define Dx 512
#define Ox 512
#define TPx (Tx + 1)            // padded time (row 0 = zeros -> causal shift)

// workspace layout (bytes)
#define XPAD_ELEMS ((long)Bx * TPx * Dx)                 // 16,781,312 bf16
#define XPAD_BYTES (XPAD_ELEMS * 2)                      // 32 MB
#define WPK_ELEMS  ((long)3 * 32 * 32 * 32 * 16)         // 1,572,864 bf16
#define WPK_BYTES  (WPK_ELEMS * 2)                       // 3 MB
#define GATE_ELEMS ((long)Bx * Tx * Ox)                  // 16,777,216 per gate

__device__ __forceinline__ float fast_sigmoid(float x) {
    return 1.0f / (1.0f + __expf(-x));
}
__device__ __forceinline__ float fast_tanh(float x) {
    return 1.0f - 2.0f / (__expf(2.0f * x) + 1.0f);
}

// --- CDNA5 async global->LDS copy (ASYNCcnt path), with portable fallback ---
__device__ __forceinline__ void async_copy_b128(const bf16* gsrc, bf16* lds_dst) {
#if __has_builtin(__builtin_amdgcn_global_load_async_to_lds_b128)
    typedef __attribute__((address_space(1))) v4i* gp_t;
    typedef __attribute__((address_space(3))) v4i* lp_t;
    __builtin_amdgcn_global_load_async_to_lds_b128(
        (gp_t)(v4i*)(bf16*)gsrc, (lp_t)(v4i*)lds_dst, 0, 0);
#else
    *(v4i*)lds_dst = *(const v4i*)gsrc;   // ld + ds_store fallback
#endif
}

__device__ __forceinline__ void wait_async0() {
#if __has_builtin(__builtin_amdgcn_s_wait_asynccnt)
    __builtin_amdgcn_s_wait_asynccnt(0);
#else
    asm volatile("s_wait_asynccnt 0x0" ::: "memory");
#endif
}

// ---------------------------------------------------------------------------
// 1) x [B,T,D] f32 -> xpad [B,T+1,D] bf16 with zeroed leading timestep.
//    K in [0,512)   reads xpad[t]   (== x[t-1])
//    K in [512,1024) reads xpad[t+1] (== x[t])
// ---------------------------------------------------------------------------
__global__ __launch_bounds__(256) void qrnn_xpad(const float* __restrict__ x,
                                                 bf16* __restrict__ xp) {
    long idx = (long)blockIdx.x * 256 + threadIdx.x;   // exact multiple of 256
    int d  = (int)(idx & 511);
    long r = idx >> 9;
    int tp = (int)(r % TPx);
    int b  = (int)(r / TPx);
    float v = (tp == 0) ? 0.0f : x[((long)b * Tx + (tp - 1)) * Dx + d];
    xp[idx] = (bf16)v;
}

// ---------------------------------------------------------------------------
// 2) Pack W_{z,f,o} [2,512,512] f32 into bf16 B-fragments:
//    wpk[g][nt][kc][lane][i]  (i = 0..15 contiguous per lane -> 32B load)
//    element i of lane L in K-chunk kc is K = kc*32 + (L>=16 ? 16 : 0) + i,
//    column o = nt*16 + (L & 15).  k -> (win = k>>9, d = k&511).
// ---------------------------------------------------------------------------
__global__ __launch_bounds__(256) void qrnn_pack_w(const float* __restrict__ Wz,
                                                   const float* __restrict__ Wf,
                                                   const float* __restrict__ Wo,
                                                   bf16* __restrict__ wpk) {
    long idx = (long)blockIdx.x * 256 + threadIdx.x;   // exact multiple of 256
    int i    = (int)(idx & 15);
    int lane = (int)((idx >> 4) & 31);
    int kc   = (int)((idx >> 9) & 31);
    int nt   = (int)((idx >> 14) & 31);
    int g    = (int)(idx >> 19);
    int k = kc * 32 + ((lane >> 4) << 4) + i;
    int w = k >> 9;
    int d = k & 511;
    int o = nt * 16 + (lane & 15);
    const float* W = (g == 0) ? Wz : ((g == 1) ? Wf : Wo);
    wpk[idx] = (bf16)W[((long)w * Dx + d) * Ox + o];
}

// ---------------------------------------------------------------------------
// 3) WMMA GEMM with LDS weight staging.
//    Block (256 thr = 8 waves) owns (b, 256-row t-strip, one 32-col n-pair).
//    All 8 waves share the same B fragments -> staged in LDS via async loads
//    in K-groups of 8 chunks (48 KB). Each wave computes TWO 16-row tiles so
//    each LDS B read feeds 2 WMMAs. Per K-chunk/wave: 2 A loads (global/L2),
//    6 B reads (LDS), 12 WMMAs. K = 1024 -> 32 chunks.
// ---------------------------------------------------------------------------
#define KG 8                                    // K-chunks per LDS stage
#define SW_ELEMS (3 * 2 * KG * 32 * 16)         // 24576 bf16 = 48 KB

__global__ __launch_bounds__(256) void qrnn_gemm(const bf16* __restrict__ xp,
                                                 const bf16* __restrict__ wpk,
                                                 const float* __restrict__ bzp,
                                                 const float* __restrict__ bfp,
                                                 const float* __restrict__ bop,
                                                 bf16* __restrict__ zb,
                                                 bf16* __restrict__ fb,
                                                 bf16* __restrict__ ob) {
    __shared__ bf16 sW[SW_ELEMS];               // [g][j][kcg][lane][16]

    int tid  = threadIdx.x;
    int wv   = tid >> 5;
    int lane = tid & 31;
    int laneHi = lane >> 4;
    int lm     = lane & 15;

    int blk = blockIdx.x;                       // 2048 blocks
    int tnp = blk & 15;                         // n-pair: cols [tnp*32, +32)
    int tmB = (blk >> 4) & 15;                  // 256-row t-strip
    int b   = blk >> 8;                         // batch

    v8f acc[3][2][2];                           // [gate][j][mt]
#pragma unroll
    for (int g = 0; g < 3; ++g)
#pragma unroll
        for (int j = 0; j < 2; ++j)
#pragma unroll
            for (int m = 0; m < 2; ++m) acc[g][j][m] = (v8f){};

    const bf16* xbase = xp + (long)b * TPx * Dx;
    int trow0 = tmB * 256 + wv * 32 + lm;       // padded row, mt adds +16

    for (int g4 = 0; g4 < 4; ++g4) {            // 4 K-groups of KG chunks
        // ---- stage this group's 6 weight fragments (48 KB) into LDS ----
        // 3072 x b128 transfers; 12 per thread. Source is contiguous per
        // (gate,j): wpk[(g*32+nt)*32 + g4*KG ...][*] = KG*512 elems.
#pragma unroll
        for (int k = 0; k < 12; ++k) {
            int c = tid + k * 256;              // transfer index
            int e = c * 8;                      // flat bf16 offset in sW
            int gj  = e >> 12;                  // 4096 elems per (g,j)
            int rem = e & 4095;
            int g = gj >> 1, j = gj & 1;
            const bf16* src = wpk +
                (((long)g * 32 + (tnp * 2 + j)) * 32 + g4 * KG) * 512 + rem;
            async_copy_b128(src, &sW[e]);
        }
        wait_async0();
        __syncthreads();

        // ---- compute on the staged group ----
#pragma unroll
        for (int kcg = 0; kcg < KG; ++kcg) {
            int kk = (g4 * KG + kcg) * 32 + laneHi * 16;
            int w  = kk >> 9;                   // conv tap (uniform per chunk)
            int d  = kk & 511;
            // A fragments (contiguous K order: 16 consecutive K per lane, matching B pack)
            v16bf a[2];
#pragma unroll
            for (int m = 0; m < 2; ++m) {
                const bf16* pa = xbase + (long)(trow0 + m * 16 + w) * Dx + d;
                v8bf alo = *(const v8bf*)(pa);
                v8bf ahi = *(const v8bf*)(pa + 8);
                a[m] = __builtin_shufflevector(alo, ahi, 0, 1, 2, 3, 4, 5, 6, 7,
                                               8, 9, 10, 11, 12, 13, 14, 15);
            }
#pragma unroll
            for (int g = 0; g < 3; ++g) {
#pragma unroll
                for (int j = 0; j < 2; ++j) {
                    v16bf bm = *(const v16bf*)
                        &sW[(((g * 2 + j) * KG + kcg) * 32 + lane) * 16];
#pragma unroll
                    for (int m = 0; m < 2; ++m) {
                        acc[g][j][m] = __builtin_amdgcn_wmma_f32_16x16x32_bf16(
                            false, a[m], false, bm, (short)0, acc[g][j][m],
                            false, false);
                    }
                }
            }
        }
        __syncthreads();                        // protect sW before next stage
    }

    // C/D layout: VGPR r, lane L -> row M = r + (L>=16)*8, col N = L&15.
#pragma unroll
    for (int j = 0; j < 2; ++j) {
        int o = (tnp * 2 + j) * 16 + lm;
        float vz = bzp[o], vf = bfp[o], vo = bop[o];
#pragma unroll
        for (int m = 0; m < 2; ++m) {
#pragma unroll
            for (int r = 0; r < 8; ++r) {
                int t = tmB * 256 + wv * 32 + m * 16 + r + laneHi * 8;
                long off = ((long)b * Tx + t) * Ox + o;
                zb[off] = (bf16)fast_tanh(acc[0][j][m][r] + vz);
                fb[off] = (bf16)fast_sigmoid(acc[1][j][m][r] + vf);
                ob[off] = (bf16)fast_sigmoid(acc[2][j][m][r] + vo);
            }
        }
    }
}

// ---------------------------------------------------------------------------
// 4) fo-pool scan: one thread per (b,o) channel, f32 recurrence over T.
//    h_t = f_t*h_{t-1} + (1-f_t)*z_t ; out = o_t * h_t
// ---------------------------------------------------------------------------
__global__ __launch_bounds__(256) void qrnn_scan(const bf16* __restrict__ zb,
                                                 const bf16* __restrict__ fb,
                                                 const bf16* __restrict__ ob,
                                                 float* __restrict__ out) {
    int tid = blockIdx.x * 256 + threadIdx.x;   // 0 .. B*O-1
    int o = tid & (Ox - 1);
    int b = tid >> 9;
    long base = (long)b * Tx * Ox + o;
    float h = 0.0f;
    for (int t = 0; t < Tx; ++t) {
        long off = base + (long)t * Ox;
        float z = (float)zb[off];
        float f = (float)fb[off];
        float g = (float)ob[off];
        h = f * h + (1.0f - f) * z;
        out[off] = g * h;
    }
}

// ---------------------------------------------------------------------------
extern "C" void kernel_launch(void* const* d_in, const int* in_sizes, int n_in,
                              void* d_out, int out_size, void* d_ws, size_t ws_size,
                              hipStream_t stream) {
    const float* x   = (const float*)d_in[0];
    const float* Wz  = (const float*)d_in[1];
    const float* Wf  = (const float*)d_in[2];
    const float* Wo  = (const float*)d_in[3];
    const float* bz  = (const float*)d_in[4];
    const float* bff = (const float*)d_in[5];
    const float* bo  = (const float*)d_in[6];

    char* ws = (char*)d_ws;
    bf16* xp  = (bf16*)(ws);
    bf16* wpk = (bf16*)(ws + XPAD_BYTES);
    bf16* zb  = (bf16*)(ws + XPAD_BYTES + WPK_BYTES);
    bf16* fbuf = zb + GATE_ELEMS;
    bf16* obuf = fbuf + GATE_ELEMS;

    // 1) pad+cast x (B*(T+1)*D elements, exact multiple of 256)
    qrnn_xpad<<<(int)(XPAD_ELEMS / 256), 256, 0, stream>>>(x, xp);
    // 2) pack weights into WMMA B-fragment layout
    qrnn_pack_w<<<(int)(WPK_ELEMS / 256), 256, 0, stream>>>(Wz, Wf, Wo, wpk);
    // 3) GEMM: blocks = B * (T/256) * (O/32) = 8*16*16 = 2048
    qrnn_gemm<<<2048, 256, 0, stream>>>(xp, wpk, bz, bff, bo, zb, fbuf, obuf);
    // 4) scan: B*O = 4096 channels
    qrnn_scan<<<(Bx * Ox) / 256, 256, 0, stream>>>(zb, fbuf, obuf, (float*)d_out);
}
